// AMRModel_48455821033670
// MI455X (gfx1250) — matrix operand order repaired and checked by reading, not compiled, not run on hardware
//
#include <hip/hip_runtime.h>
#include <stddef.h>

// ---------------------------------------------------------------------------
// Model dims (from reference)
// ---------------------------------------------------------------------------
#define B_ 8
#define T_ 16
#define S_ 64
#define NPT_ 24
#define NL_ 16
#define EPT_ 48
#define H_ 768
#define NH_ 12
#define DH_ 64
#define FF_ 3072
#define R_ 16
#define NB_ 30
#define H1_ 256
#define H2_ 256
#define CLF_ 768
#define LBL_ 7
#define BT_ (B_ * T_)              // 128
#define NN_ (BT_ * NPT_)           // 3072 graph nodes
#define E_ (BT_ * EPT_)            // 6144 edges
#define NODE_CHUNK_SEQS 512        // 512 node-seqs * 16 tok = 8192 tokens/chunk

typedef __attribute__((ext_vector_type(16))) _Float16 v16h;
typedef __attribute__((ext_vector_type(8)))  float    v8f;

// ---------------------------------------------------------------------------
// Helpers
// ---------------------------------------------------------------------------
__device__ __forceinline__ float block_reduce_sum(float v, float* sh) {
  int t = threadIdx.x;
  sh[t] = v;
  __syncthreads();
  for (int s = blockDim.x >> 1; s > 0; s >>= 1) {
    if (t < s) sh[t] += sh[t + s];
    __syncthreads();
  }
  float r = sh[0];
  __syncthreads();
  return r;
}

__device__ __forceinline__ void atomicMaxFloat(float* addr, float val) {
  unsigned int* ua = (unsigned int*)addr;
  unsigned int old = *ua;
  while (true) {
    float f = __uint_as_float(old);
    if (f >= val) break;
    unsigned int assumed = old;
    old = atomicCAS(ua, assumed, __float_as_uint(val));
    if (old == assumed) break;
  }
}

__device__ __forceinline__ float gelu_tanh(float x) {
  float t = 0.7978845608028654f * (x + 0.044715f * x * x * x);
  return 0.5f * x * (1.0f + tanhf(t));
}

// ---------------------------------------------------------------------------
// WMMA fragment loaders (ISA 7.12.2 layouts), f32 memory -> f16 fragments.
// ---------------------------------------------------------------------------
// A fragment (16x32 f16): lane half (lane>>4) selects K-octet;
//   VGPR v pairs (2v,2v+1): half0 K=2v, half1 K=8+2v ; VGPR4-7: +16.
__device__ __forceinline__ v16h load_a_frag(const float* __restrict__ Arow,
                                            int k0, int half) {
  v16h a{};
#pragma unroll
  for (int v = 0; v < 4; ++v) {
    int kk = k0 + half * 8 + 2 * v;
    a[2 * v]     = (_Float16)Arow[kk];
    a[2 * v + 1] = (_Float16)Arow[kk + 1];
  }
#pragma unroll
  for (int v = 0; v < 4; ++v) {
    int kk = k0 + 16 + half * 8 + 2 * v;
    a[8 + 2 * v] = (_Float16)Arow[kk];
    a[9 + 2 * v] = (_Float16)Arow[kk + 1];
  }
  return a;
}

// B fragment (32x16 f16): same K order as A: VGPR v (0-3) pairs half0 K=2v,
//   half1 K=8+2v ; VGPR4-7: +16.
__device__ __forceinline__ v16h load_b_frag(const float* __restrict__ Bcol,
                                            int k0, int half, int N) {
  v16h b{};
#pragma unroll
  for (int v = 0; v < 8; ++v) {
    int kk = k0 + (v >> 2) * 16 + half * 8 + 2 * (v & 3);
    b[2 * v]     = (_Float16)Bcol[(size_t)kk * N];
    b[2 * v + 1] = (_Float16)Bcol[(size_t)(kk + 1) * N];
  }
  return b;
}

// ---------------------------------------------------------------------------
// WMMA GEMM: C[M,N] = act(A[M,K] @ B[K,N] + bias), f32 in memory, f16
// fragments, f32 accumulation.  One wave per 32x32 tile (2x2 register
// blocking -> 4 WMMA issues / K-step, 2x arithmetic intensity).
// Requires M%32==0, N%32==0, K%32==0 (true for all GEMMs in this model).
// act: 0=none, 1=gelu, 2=relu
// ---------------------------------------------------------------------------
__global__ void __launch_bounds__(32)
gemm_wmma_kernel(const float* __restrict__ A, const float* __restrict__ Bm,
                 const float* __restrict__ bias, float* __restrict__ C,
                 int M, int N, int K, int act) {
  const int tm = blockIdx.x;      // 32-row tile
  const int tn = blockIdx.y;      // 32-col tile
  const int lane = threadIdx.x;
  const int half = lane >> 4;     // 0: lanes 0-15, 1: lanes 16-31
  const int idx  = lane & 15;

  const float* Arow0 = A  + (size_t)(tm * 32 + idx) * K;        // rows idx
  const float* Arow1 = Arow0 + (size_t)16 * K;                  // rows idx+16
  const float* Bcol0 = Bm + (size_t)(tn * 32 + idx);            // cols idx
  const float* Bcol1 = Bcol0 + 16;                              // cols idx+16

  v8f acc00 = {}, acc01 = {}, acc10 = {}, acc11 = {};
  for (int k0 = 0; k0 < K; k0 += 32) {
    if (k0 + 64 < K) {
      __builtin_prefetch(Arow0 + k0 + 64, 0, 0);
      __builtin_prefetch(Arow1 + k0 + 64, 0, 0);
      __builtin_prefetch(Bcol0 + (size_t)(k0 + 64) * N, 0, 0);
    }
    v16h a0 = load_a_frag(Arow0, k0, half);
    v16h a1 = load_a_frag(Arow1, k0, half);
    v16h b0 = load_b_frag(Bcol0, k0, half, N);
    v16h b1 = load_b_frag(Bcol1, k0, half, N);
    acc00 = __builtin_amdgcn_wmma_f32_16x16x32_f16(false, a0, false, b0,
                                                   (short)0, acc00, false, false);
    acc01 = __builtin_amdgcn_wmma_f32_16x16x32_f16(false, a0, false, b1,
                                                   (short)0, acc01, false, false);
    acc10 = __builtin_amdgcn_wmma_f32_16x16x32_f16(false, a1, false, b0,
                                                   (short)0, acc10, false, false);
    acc11 = __builtin_amdgcn_wmma_f32_16x16x32_f16(false, a1, false, b1,
                                                   (short)0, acc11, false, false);
  }

  const int col0 = tn * 32 + idx;
  const int col1 = col0 + 16;
  const float bv0 = bias ? bias[col0] : 0.0f;
  const float bv1 = bias ? bias[col1] : 0.0f;
#pragma unroll
  for (int i = 0; i < 8; ++i) {
    int row0 = tm * 32 + i + half * 8;   // C/D layout: VGPR i -> M = i + 8*half
    int row1 = row0 + 16;
    float x00 = acc00[i] + bv0, x01 = acc01[i] + bv1;
    float x10 = acc10[i] + bv0, x11 = acc11[i] + bv1;
    if (act == 1) {
      x00 = gelu_tanh(x00); x01 = gelu_tanh(x01);
      x10 = gelu_tanh(x10); x11 = gelu_tanh(x11);
    } else if (act == 2) {
      x00 = fmaxf(x00, 0.0f); x01 = fmaxf(x01, 0.0f);
      x10 = fmaxf(x10, 0.0f); x11 = fmaxf(x11, 0.0f);
    }
    C[(size_t)row0 * N + col0] = x00;
    C[(size_t)row0 * N + col1] = x01;
    C[(size_t)row1 * N + col0] = x10;
    C[(size_t)row1 * N + col1] = x11;
  }
}

// ---------------------------------------------------------------------------
// Embedding gather + positional + LayerNorm.  One block per token, 256 thr,
// H = 768 => 3 elements / thread.
// ---------------------------------------------------------------------------
__global__ void embed_ln_kernel(const int* __restrict__ ids,
                                const float* __restrict__ emb,
                                const float* __restrict__ pos,
                                const float* __restrict__ g,
                                const float* __restrict__ b,
                                float* __restrict__ out, int L) {
  __shared__ float sh[256];
  const int tok = blockIdx.x;
  const int l = tok % L;
  const int id = ids[tok];
  const float* e = emb + (size_t)id * H_;
  const float* p = pos + (size_t)l * H_;
  float vals[3];
  float s = 0.f, s2 = 0.f;
#pragma unroll
  for (int j = 0; j < 3; ++j) {
    int i = threadIdx.x + j * 256;
    float x = e[i] + p[i];
    vals[j] = x; s += x; s2 += x * x;
  }
  float sum  = block_reduce_sum(s, sh);
  float sum2 = block_reduce_sum(s2, sh);
  float mu  = sum / H_;
  float var = sum2 / H_ - mu * mu;
  float inv = rsqrtf(var + 1e-12f);
#pragma unroll
  for (int j = 0; j < 3; ++j) {
    int i = threadIdx.x + j * 256;
    out[(size_t)tok * H_ + i] = (vals[j] - mu) * inv * g[i] + b[i];
  }
}

// out = LN(a + b2), H=768
__global__ void add_ln_kernel(const float* __restrict__ a,
                              const float* __restrict__ b2,
                              const float* __restrict__ g,
                              const float* __restrict__ bb,
                              float* __restrict__ out) {
  __shared__ float sh[256];
  const int tok = blockIdx.x;
  float vals[3];
  float s = 0.f, s2 = 0.f;
#pragma unroll
  for (int j = 0; j < 3; ++j) {
    int i = threadIdx.x + j * 256;
    float x = a[(size_t)tok * H_ + i] + b2[(size_t)tok * H_ + i];
    vals[j] = x; s += x; s2 += x * x;
  }
  float sum  = block_reduce_sum(s, sh);
  float sum2 = block_reduce_sum(s2, sh);
  float mu  = sum / H_;
  float var = sum2 / H_ - mu * mu;
  float inv = rsqrtf(var + 1e-12f);
#pragma unroll
  for (int j = 0; j < 3; ++j) {
    int i = threadIdx.x + j * 256;
    out[(size_t)tok * H_ + i] = (vals[j] - mu) * inv * g[i] + bb[i];
  }
}

// ---------------------------------------------------------------------------
// Multi-head attention.  grid=(nseq, NH), block=64.  L <= 64, dh = 64.
// ---------------------------------------------------------------------------
__global__ void __launch_bounds__(64)
attn_kernel(const float* __restrict__ q, const float* __restrict__ k,
            const float* __restrict__ v, const int* __restrict__ mask,
            float* __restrict__ ctx, int L) {
  __shared__ float Qs[64 * 64], Ks[64 * 64], Vs[64 * 64];
  const int seq = blockIdx.x;
  const int h = blockIdx.y;
  const int t = threadIdx.x;       // 0..63 : column loader / query row
  for (int l = 0; l < L; ++l) {
    size_t base = ((size_t)(seq * L + l)) * H_ + h * DH_ + t;
    Qs[l * 64 + t] = q[base];
    Ks[l * 64 + t] = k[base];
    Vs[l * 64 + t] = v[base];
  }
  __syncthreads();
  if (t < L) {
    float s[64];
    float mx = -1e30f;
    for (int j = 0; j < L; ++j) {
      float acc = 0.f;
#pragma unroll 8
      for (int d = 0; d < 64; ++d) acc += Qs[t * 64 + d] * Ks[j * 64 + d];
      acc *= 0.125f;                             // 1/sqrt(64)
      if (mask[seq * L + j] == 0) acc += -1e9f;
      s[j] = acc;
      mx = fmaxf(mx, acc);
    }
    float den = 0.f;
    for (int j = 0; j < L; ++j) { s[j] = __expf(s[j] - mx); den += s[j]; }
    float invd = 1.0f / den;
    for (int d = 0; d < 64; ++d) {
      float c = 0.f;
      for (int j = 0; j < L; ++j) c += s[j] * Vs[j * 64 + d];
      ctx[((size_t)(seq * L + t)) * H_ + h * DH_ + d] = c * invd;
    }
  }
}

// CLS pooling (token 0 of each utterance sequence)
__global__ void cls_copy_kernel(const float* __restrict__ x, float* __restrict__ out, int L) {
  const int seq = blockIdx.x;
#pragma unroll
  for (int j = 0; j < 3; ++j) {
    int i = threadIdx.x + j * 256;
    out[(size_t)seq * H_ + i] = x[(size_t)seq * L * H_ + i];
  }
}

// Attention pooling over NL node tokens.  One block / node, 256 threads.
__global__ void attnpool_kernel(const float* __restrict__ hn,
                                const int* __restrict__ nmask,
                                const float* __restrict__ pw, const float* __restrict__ pb,
                                float* __restrict__ x0) {
  __shared__ float sh[256];
  __shared__ float sc[NL_];
  const int node = blockIdx.x;     // chunk-local
  for (int tt = 0; tt < NL_; ++tt) {
    float part = 0.f;
#pragma unroll
    for (int j = 0; j < 3; ++j) {
      int i = threadIdx.x + j * 256;
      part += hn[((size_t)(node * NL_ + tt)) * H_ + i] * pw[i];
    }
    float tot = block_reduce_sum(part, sh);
    if (threadIdx.x == 0) sc[tt] = tot + pb[0];
    __syncthreads();
  }
  if (threadIdx.x == 0) {
    float mx = -1e30f;
    for (int tt = 0; tt < NL_; ++tt) {
      float sv = (nmask[node * NL_ + tt] > 0) ? sc[tt] : -1e9f;
      sc[tt] = sv;
      mx = fmaxf(mx, sv);
    }
    float den = 0.f;
    for (int tt = 0; tt < NL_; ++tt) { sc[tt] = __expf(sc[tt] - mx); den += sc[tt]; }
    float inv = 1.0f / den;
    for (int tt = 0; tt < NL_; ++tt) sc[tt] *= inv;
  }
  __syncthreads();
#pragma unroll
  for (int j = 0; j < 3; ++j) {
    int i = threadIdx.x + j * 256;
    float acc = 0.f;
    for (int tt = 0; tt < NL_; ++tt)
      acc += sc[tt] * hn[((size_t)(node * NL_ + tt)) * H_ + i];
    x0[(size_t)node * H_ + i] = acc;
  }
}

// ---------------------------------------------------------------------------
// R-GAT kernels
// ---------------------------------------------------------------------------
// W[r] = sum_b comp[r,b] * basis[b]   (out: [R, IN, OUT])
__global__ void relw_kernel(const float* __restrict__ basis,
                            const float* __restrict__ comp,
                            float* __restrict__ W, int IN, int OUT) {
  const size_t total = (size_t)R_ * IN * OUT;
  for (size_t i = (size_t)blockIdx.x * blockDim.x + threadIdx.x; i < total;
       i += (size_t)gridDim.x * blockDim.x) {
    size_t io = i % ((size_t)IN * OUT);
    int r = (int)(i / ((size_t)IN * OUT));
    float acc = 0.f;
    for (int bb = 0; bb < NB_; ++bb)
      acc += comp[r * NB_ + bb] * basis[(size_t)bb * IN * OUT + io];
    W[i] = acc;
  }
}

__global__ void fill_kernel(float* __restrict__ p, float v, int n) {
  int i = blockIdx.x * 256 + threadIdx.x;
  if (i < n) p[i] = v;
}

// Per-edge message + attention logit. block=OUT(=256), grid=E.
__global__ void __launch_bounds__(256)
edge_msg_kernel(const float* __restrict__ xin, const float* __restrict__ Wrel,
                const int* __restrict__ eidx, const int* __restrict__ etype,
                const float* __restrict__ aq, const float* __restrict__ ak,
                float* __restrict__ msg, float* __restrict__ escore, int IN) {
  __shared__ float sh[256];
  const int e = blockIdx.x;
  const int o = threadIdx.x;
  const int bt = e / EPT_;
  const int s = eidx[e * 2 + 0] + bt * NPT_;
  const int d = eidx[e * 2 + 1] + bt * NPT_;
  const int r = etype[e];
  const float* W  = Wrel + (size_t)r * IN * H1_;
  const float* xs = xin + (size_t)s * IN;
  const float* xd = xin + (size_t)d * IN;
  float ms = 0.f, qs = 0.f;
  for (int i = 0; i < IN; ++i) {
    float w = W[(size_t)i * H1_ + o];
    ms += xs[i] * w;
    qs += xd[i] * w;
  }
  msg[(size_t)e * H1_ + o] = ms;
  float tot = block_reduce_sum(qs * aq[o] + ms * ak[o], sh);
  if (o == 0) escore[e] = (tot > 0.f) ? tot : 0.2f * tot;   // leaky_relu(0.2)
}

__global__ void seg_max_kernel(const float* __restrict__ escore,
                               const int* __restrict__ eidx,
                               float* __restrict__ mmax) {
  int e = blockIdx.x * 256 + threadIdx.x;
  if (e < E_) {
    int d = eidx[e * 2 + 1] + (e / EPT_) * NPT_;
    atomicMaxFloat(&mmax[d], escore[e]);
  }
}

__global__ void seg_expsum_kernel(const float* __restrict__ escore,
                                  const int* __restrict__ eidx,
                                  const float* __restrict__ mmax,
                                  float* __restrict__ exv, float* __restrict__ den) {
  int e = blockIdx.x * 256 + threadIdx.x;
  if (e < E_) {
    int d = eidx[e * 2 + 1] + (e / EPT_) * NPT_;
    float ex = __expf(escore[e] - mmax[d]);
    exv[e] = ex;
    atomicAdd(&den[d], ex);
  }
}

__global__ void __launch_bounds__(256)
edge_scatter_kernel(const float* __restrict__ msg, const float* __restrict__ exv,
                    const float* __restrict__ den, const int* __restrict__ eidx,
                    float* __restrict__ out) {
  const int e = blockIdx.x;
  const int o = threadIdx.x;
  const int d = eidx[e * 2 + 1] + (e / EPT_) * NPT_;
  float alpha = exv[e] / den[d];
  atomicAdd(&out[(size_t)d * H1_ + o], alpha * msg[(size_t)e * H1_ + o]);
}

// mean over NPT nodes per utterance
__global__ void meanpool_kernel(const float* __restrict__ x2, float* __restrict__ inner) {
  int i = blockIdx.x * 256 + threadIdx.x;
  if (i < BT_ * H2_) {
    int bt = i / H2_, o = i % H2_;
    float s = 0.f;
    for (int p = 0; p < NPT_; ++p) s += x2[((size_t)(bt * NPT_ + p)) * H2_ + o];
    inner[i] = s * (1.0f / NPT_);
  }
}

__global__ void concat_kernel(const float* __restrict__ utt,
                              const float* __restrict__ inner,
                              float* __restrict__ feat) {
  int i = blockIdx.x * 256 + threadIdx.x;
  if (i < BT_ * (H_ + H2_)) {
    int bt = i / (H_ + H2_), c = i % (H_ + H2_);
    feat[i] = (c < H_) ? utt[(size_t)bt * H_ + c] : inner[(size_t)bt * H2_ + (c - H_)];
  }
}

__global__ void logits_kernel(const float* __restrict__ z, const float* __restrict__ w,
                              const float* __restrict__ b, float* __restrict__ out) {
  int i = blockIdx.x * 256 + threadIdx.x;
  if (i < BT_ * LBL_) {
    int m = i / LBL_, n = i % LBL_;
    float s = b[n];
    for (int kk = 0; kk < CLF_; ++kk) s += z[(size_t)m * CLF_ + kk] * w[kk * LBL_ + n];
    out[i] = s;
  }
}

__global__ void __launch_bounds__(128)
loss_kernel(const float* __restrict__ logits, const int* __restrict__ lbl,
            float* __restrict__ out) {
  __shared__ float red[128];
  int t = threadIdx.x;          // 128 rows
  float mx = -1e30f;
  for (int j = 0; j < LBL_; ++j) mx = fmaxf(mx, logits[t * LBL_ + j]);
  float se = 0.f;
  for (int j = 0; j < LBL_; ++j) se += __expf(logits[t * LBL_ + j] - mx);
  float lse = mx + __logf(se);
  red[t] = -(logits[t * LBL_ + lbl[t]] - lse);
  __syncthreads();
  for (int s = 64; s > 0; s >>= 1) {
    if (t < s) red[t] += red[t + s];
    __syncthreads();
  }
  if (t == 0) out[0] = red[0] / (float)BT_;
}

// ---------------------------------------------------------------------------
// Host orchestration
// ---------------------------------------------------------------------------
struct EncP {
  const float *emb, *pos, *wq, *wk, *wv, *wo, *bq, *bk, *bv, *bo;
  const float *w1, *bf1, *w2, *bf2, *g0, *b0, *g1, *b1, *g2, *b2;
};

static EncP make_enc_p(void* const* d, int b) {
  EncP p;
  p.emb = (const float*)d[b + 0];  p.pos = (const float*)d[b + 1];
  p.wq  = (const float*)d[b + 2];  p.wk  = (const float*)d[b + 3];
  p.wv  = (const float*)d[b + 4];  p.wo  = (const float*)d[b + 5];
  p.bq  = (const float*)d[b + 6];  p.bk  = (const float*)d[b + 7];
  p.bv  = (const float*)d[b + 8];  p.bo  = (const float*)d[b + 9];
  p.w1  = (const float*)d[b + 10]; p.bf1 = (const float*)d[b + 11];
  p.w2  = (const float*)d[b + 12]; p.bf2 = (const float*)d[b + 13];
  p.g0  = (const float*)d[b + 14]; p.b0  = (const float*)d[b + 15];
  p.g1  = (const float*)d[b + 16]; p.b1  = (const float*)d[b + 17];
  p.g2  = (const float*)d[b + 18]; p.b2  = (const float*)d[b + 19];
  return p;
}

static void run_encoder(hipStream_t stream, const EncP& P,
                        const int* ids, const int* mask, int nseq, int L,
                        float* xbuf, float* qbuf, float* kbuf, float* vbuf,
                        float* cbuf, float* fbuf) {
  const int Ttok = nseq * L;                       // multiple of 32 here
  embed_ln_kernel<<<Ttok, 256, 0, stream>>>(ids, P.emb, P.pos, P.g0, P.b0, xbuf, L);
  dim3 gH(Ttok / 32, H_ / 32);
  gemm_wmma_kernel<<<gH, 32, 0, stream>>>(xbuf, P.wq, P.bq, qbuf, Ttok, H_, H_, 0);
  gemm_wmma_kernel<<<gH, 32, 0, stream>>>(xbuf, P.wk, P.bk, kbuf, Ttok, H_, H_, 0);
  gemm_wmma_kernel<<<gH, 32, 0, stream>>>(xbuf, P.wv, P.bv, vbuf, Ttok, H_, H_, 0);
  attn_kernel<<<dim3(nseq, NH_), 64, 0, stream>>>(qbuf, kbuf, vbuf, mask, cbuf, L);
  gemm_wmma_kernel<<<gH, 32, 0, stream>>>(cbuf, P.wo, P.bo, vbuf, Ttok, H_, H_, 0);
  add_ln_kernel<<<Ttok, 256, 0, stream>>>(xbuf, vbuf, P.g1, P.b1, qbuf);   // qbuf = x1
  gemm_wmma_kernel<<<dim3(Ttok / 32, FF_ / 32), 32, 0, stream>>>(
      qbuf, P.w1, P.bf1, fbuf, Ttok, FF_, H_, 1 /*gelu*/);
  gemm_wmma_kernel<<<gH, 32, 0, stream>>>(fbuf, P.w2, P.bf2, kbuf, Ttok, H_, FF_, 0);
  add_ln_kernel<<<Ttok, 256, 0, stream>>>(qbuf, kbuf, P.g2, P.b2, xbuf);   // xbuf = out
}

extern "C" void kernel_launch(void* const* d_in, const int* in_sizes, int n_in,
                              void* d_out, int out_size, void* d_ws, size_t ws_size,
                              hipStream_t stream) {
  (void)in_sizes; (void)n_in; (void)out_size; (void)ws_size;
  // ---- input indices (setup_inputs dict order, params flattened in insertion order) ----
  const int* ids    = (const int*)d_in[0];   // [B,T,S]
  const int* amask  = (const int*)d_in[1];
  const int* nids   = (const int*)d_in[2];   // [B,T,NPT,NL]
  const int* nmask  = (const int*)d_in[3];
  const int* eidx   = (const int*)d_in[4];   // [B,T,EPT,2]
  const int* etype  = (const int*)d_in[5];
  const int* lbl    = (const int*)d_in[6];
  EncP uttP  = make_enc_p(d_in, 7);
  EncP nodeP = make_enc_p(d_in, 27);
  const float* pool_w = (const float*)d_in[47];
  const float* pool_b = (const float*)d_in[48];
  const float* basis1 = (const float*)d_in[49];
  const float* comp1  = (const float*)d_in[50];
  const float* aq1    = (const float*)d_in[51];
  const float* ak1    = (const float*)d_in[52];
  const float* basis2 = (const float*)d_in[53];
  const float* comp2  = (const float*)d_in[54];
  const float* aq2    = (const float*)d_in[55];
  const float* ak2    = (const float*)d_in[56];
  const float* clf_w1 = (const float*)d_in[57];
  const float* clf_b1 = (const float*)d_in[58];
  const float* clf_w2 = (const float*)d_in[59];
  const float* clf_b2 = (const float*)d_in[60];
  const float* clf_w3 = (const float*)d_in[61];
  const float* clf_b3 = (const float*)d_in[62];

  // ---- workspace layout (floats) ----
  const size_t SC = 8192ull * H_;              // encoder chunk token-buffer
  float* ws = (float*)d_ws;
  float* xbuf = ws;
  float* qbuf = xbuf + SC;
  float* kbuf = qbuf + SC;
  float* vbuf = kbuf + SC;
  float* cbuf = vbuf + SC;
  float* fbuf = cbuf + SC;                     // 8192 * FF
  float* utt_cls = fbuf + 8192ull * FF_;       // [128, 768]
  float* x0   = utt_cls + (size_t)BT_ * H_;    // [3072, 768]
  float* W1rel = x0 + (size_t)NN_ * H_;        // [16, 768, 256]
  float* W2rel = W1rel + (size_t)R_ * H_ * H1_;// [16, 256, 256]
  float* msg  = W2rel + (size_t)R_ * H1_ * H2_;// [6144, 256]
  float* escore = msg + (size_t)E_ * H1_;
  float* exv  = escore + E_;
  float* mmax = exv + E_;
  float* den  = mmax + NN_;
  float* x1g  = den + NN_;                     // [3072, 256]
  float* x2g  = x1g + (size_t)NN_ * H1_;       // [3072, 256]
  float* inner = x2g + (size_t)NN_ * H2_;      // [128, 256]
  float* feat = inner + (size_t)BT_ * H2_;     // [128, 1024]
  float* z1   = feat + (size_t)BT_ * (H_ + H2_);
  float* z2   = z1 + (size_t)BT_ * CLF_;

  // ---- 1) utterance encoder (128 seqs x 64 tokens), CLS pool ----
  run_encoder(stream, uttP, ids, amask, BT_, S_, xbuf, qbuf, kbuf, vbuf, cbuf, fbuf);
  cls_copy_kernel<<<BT_, 256, 0, stream>>>(xbuf, utt_cls, S_);

  // ---- 2) node encoder, chunked (6 x 512 node-seqs), attention pool per chunk ----
  for (int c = 0; c < NN_ / NODE_CHUNK_SEQS; ++c) {
    const int* cids = nids + (size_t)c * NODE_CHUNK_SEQS * NL_;
    const int* cmsk = nmask + (size_t)c * NODE_CHUNK_SEQS * NL_;
    run_encoder(stream, nodeP, cids, cmsk, NODE_CHUNK_SEQS, NL_,
                xbuf, qbuf, kbuf, vbuf, cbuf, fbuf);
    attnpool_kernel<<<NODE_CHUNK_SEQS, 256, 0, stream>>>(
        xbuf, cmsk, pool_w, pool_b, x0 + (size_t)c * NODE_CHUNK_SEQS * H_);
  }

  // ---- 3) R-GAT conv1 (768 -> 256) ----
  relw_kernel<<<2048, 256, 0, stream>>>(basis1, comp1, W1rel, H_, H1_);
  fill_kernel<<<(NN_ + 255) / 256, 256, 0, stream>>>(mmax, -1e30f, NN_);
  fill_kernel<<<(NN_ + 255) / 256, 256, 0, stream>>>(den, 0.f, NN_);
  fill_kernel<<<(NN_ * H1_ + 255) / 256, 256, 0, stream>>>(x1g, 0.f, NN_ * H1_);
  edge_msg_kernel<<<E_, 256, 0, stream>>>(x0, W1rel, eidx, etype, aq1, ak1, msg, escore, H_);
  seg_max_kernel<<<(E_ + 255) / 256, 256, 0, stream>>>(escore, eidx, mmax);
  seg_expsum_kernel<<<(E_ + 255) / 256, 256, 0, stream>>>(escore, eidx, mmax, exv, den);
  edge_scatter_kernel<<<E_, 256, 0, stream>>>(msg, exv, den, eidx, x1g);

  // ---- 4) R-GAT conv2 (256 -> 256) ----
  relw_kernel<<<2048, 256, 0, stream>>>(basis2, comp2, W2rel, H1_, H2_);
  fill_kernel<<<(NN_ + 255) / 256, 256, 0, stream>>>(mmax, -1e30f, NN_);
  fill_kernel<<<(NN_ + 255) / 256, 256, 0, stream>>>(den, 0.f, NN_);
  fill_kernel<<<(NN_ * H2_ + 255) / 256, 256, 0, stream>>>(x2g, 0.f, NN_ * H2_);
  edge_msg_kernel<<<E_, 256, 0, stream>>>(x1g, W2rel, eidx, etype, aq2, ak2, msg, escore, H1_);
  seg_max_kernel<<<(E_ + 255) / 256, 256, 0, stream>>>(escore, eidx, mmax);
  seg_expsum_kernel<<<(E_ + 255) / 256, 256, 0, stream>>>(escore, eidx, mmax, exv, den);
  edge_scatter_kernel<<<E_, 256, 0, stream>>>(msg, exv, den, eidx, x2g);

  // ---- 5) pooling, concat, classifier MLP (WMMA), loss ----
  meanpool_kernel<<<(BT_ * H2_ + 255) / 256, 256, 0, stream>>>(x2g, inner);
  concat_kernel<<<(BT_ * (H_ + H2_) + 255) / 256, 256, 0, stream>>>(utt_cls, inner, feat);
  gemm_wmma_kernel<<<dim3(BT_ / 32, CLF_ / 32), 32, 0, stream>>>(
      feat, clf_w1, clf_b1, z1, BT_, CLF_, H_ + H2_, 2 /*relu*/);
  gemm_wmma_kernel<<<dim3(BT_ / 32, CLF_ / 32), 32, 0, stream>>>(
      z1, clf_w2, clf_b2, z2, BT_, CLF_, CLF_, 2 /*relu*/);
  float* out_logits = (float*)d_out;
  logits_kernel<<<(BT_ * LBL_ + 255) / 256, 256, 0, stream>>>(z2, clf_w3, clf_b3, out_logits);
  loss_kernel<<<1, 128, 0, stream>>>(out_logits, lbl, out_logits + BT_ * LBL_);
}
